// QuantizedLinear_29394756174133
// MI455X (gfx1250) — matrix operand order repaired and checked by reading, compiled, not run on hardware
//
#include <hip/hip_runtime.h>
#include <stdint.h>

#define OUT_N  11008
#define IN_K   4096
#define M_ROWS 8192

#define BM 128
#define BN 128
#define BK 64
#define LDT 72              // padded LDS row stride in bf16 halfwords (144 B)
#define PK_ROW (IN_K / 4)   // 1024 packed int32 per weight row

typedef __attribute__((ext_vector_type(16))) __bf16 v16bf;
typedef __attribute__((ext_vector_type(8)))  __bf16 v8bf;
typedef __attribute__((ext_vector_type(8)))  float  v8f;

__device__ __forceinline__ unsigned short f2bf(float f) {
    __bf16 h = (__bf16)f;                     // native bf16 convert path
    return __builtin_bit_cast(unsigned short, h);
}

__device__ __forceinline__ unsigned short tern_bits(unsigned t) {
    // t in {0,1,2} encodes {-1, 0, +1} as bf16 bit patterns
    return (t == 0u) ? (unsigned short)0xBF80u
         : ((t == 2u) ? (unsigned short)0x3F80u : (unsigned short)0x0000u);
}

// Issue async copy of one 8 KB packed-weight K-tile (128 rows x 64 B) into LDS.
// 256 threads x 2 x 16 B chunks; saddr = packed base, vaddr = per-lane byte offset.
__device__ __forceinline__ void issue_async_B(uint64_t pbase, int blockN0, int kt,
                                              unsigned ldsbase, int tid) {
    #pragma unroll
    for (int i = 0; i < 2; ++i) {
        int idx = tid + i * 256;              // 0..511 chunk id
        int n   = idx >> 2;                   // weight row 0..127
        int ch  = idx & 3;                    // 16 B chunk within 64 B row
        unsigned goff = ((unsigned)(blockN0 + n) * (unsigned)PK_ROW
                         + (unsigned)(kt >> 2)) * 4u + (unsigned)ch * 16u;
        unsigned laddr = ldsbase + (unsigned)(n * 64 + ch * 16);
        asm volatile("global_load_async_to_lds_b128 %0, %1, %2"
                     :: "v"(laddr), "v"(goff), "s"(pbase)
                     : "memory");
    }
}

__global__ __launch_bounds__(256)
void ternary_gemm_bf16_wmma(const float* __restrict__ x,
                            const int*   __restrict__ packed,
                            const float* __restrict__ scale,
                            float*       __restrict__ out) {
    __shared__ unsigned short As[BM * LDT];   // A tile [m][k] bf16
    __shared__ unsigned short Bs[BN * LDT];   // B tile [n][k] bf16
    __shared__ int   Braw[2][BN * 16];        // double-buffered raw packed weights
    __shared__ uint2 lut[256];                // byte -> 4 bf16 ternary values

    const int tid   = threadIdx.x;
    const int lane  = tid & 31;
    const int wave  = tid >> 5;
    const int waveM = wave & 1;     // 2 waves along M (64 rows each)
    const int waveN = wave >> 1;    // 4 waves along N (32 cols each)
    const int l15   = lane & 15;
    const int hi16  = lane >> 4;

    const int blockN0 = blockIdx.x * BN;
    const int blockM0 = blockIdx.y * BM;

    // ---- one-time LUT init: byte (4 x 2-bit) -> 4 bf16 halfwords (8 B) ----
    {
        unsigned w = (unsigned)tid;           // exactly 256 threads
        uint2 e;
        e.x = (unsigned)tern_bits(w & 3u)        | ((unsigned)tern_bits((w >> 2) & 3u) << 16);
        e.y = (unsigned)tern_bits((w >> 4) & 3u) | ((unsigned)tern_bits((w >> 6) & 3u) << 16);
        lut[tid] = e;
    }

    const uint64_t pbase = (uint64_t)(uintptr_t)packed;
    const unsigned braw0 = (unsigned)(uintptr_t)(&Braw[0][0]);  // LDS byte offset
    const unsigned braw1 = (unsigned)(uintptr_t)(&Braw[1][0]);

    v8f acc[4][2];
    #pragma unroll
    for (int mi = 0; mi < 4; ++mi)
        #pragma unroll
        for (int ni = 0; ni < 2; ++ni) { v8f z = {}; acc[mi][ni] = z; }

    // prime the async pipeline with tile 0
    issue_async_B(pbase, blockN0, 0, braw0, tid);

    for (int kt = 0; kt < IN_K; kt += BK) {
        const int buf = (kt / BK) & 1;

        // this wave's async copies for tile kt are complete...
        asm volatile("s_wait_asynccnt 0" ::: "memory");
        // ...and after the barrier, every wave's are (also fences prev compute)
        __syncthreads();

        // ---- stage A: 128 x 64 fp32 -> bf16 into LDS (8 float4 / thread) ----
        {
            const float4* xsrc = (const float4*)(x + (size_t)blockM0 * IN_K + kt);
            #pragma unroll
            for (int i = 0; i < 8; ++i) {
                int idx = tid + i * 256;      // 0..2047 float4s
                int r   = idx >> 4;           // row 0..127
                int c4  = idx & 15;           // float4 within 64-wide row
                float4 v = xsrc[(size_t)r * (IN_K / 4) + c4];
                uint2 p;
                p.x = (unsigned)f2bf(v.x) | ((unsigned)f2bf(v.y) << 16);
                p.y = (unsigned)f2bf(v.z) | ((unsigned)f2bf(v.w) << 16);
                *(uint2*)(&As[r * LDT + c4 * 4]) = p;
            }
        }

        // ---- stage B: unpack async-staged packed tile via LDS LUT ----
        {
            const int n  = tid >> 1;          // weight row 0..127
            const int c0 = (tid & 1) * 8;     // first of 8 int32s in this row
            const uint4* bp = (const uint4*)(&Braw[buf][n * 16 + c0]);
            uint4 w0 = bp[0];
            uint4 w1 = bp[1];
            unsigned wv[8] = { w0.x, w0.y, w0.z, w0.w, w1.x, w1.y, w1.z, w1.w };
            #pragma unroll
            for (int j = 0; j < 8; ++j) {
                uint2 e = lut[wv[j] & 255u];
                *(uint2*)(&Bs[n * LDT + (c0 + j) * 4]) = e;
            }
        }

        // kick off async copy of the NEXT packed tile; overlaps with compute
        if (kt + BK < IN_K) {
            issue_async_B(pbase, blockN0, kt + BK, (buf ? braw0 : braw1), tid);
        }

        __syncthreads();

        // prefetch next A tile (global_prefetch_b8) while computing
        if (kt + BK < IN_K) {
            const float* xn = x + (size_t)(blockM0 + (tid & 127)) * IN_K
                                + (kt + BK) + ((tid >> 7) << 5);
            __builtin_prefetch(xn, 0, 1);
        }

        // ---- compute: 2 WMMA K-steps of 32 ----
        #pragma unroll
        for (int kk = 0; kk < BK; kk += 32) {
            // A frags: lanes 0-15 K 0-7 / 16-23 ; lanes 16-31 K 8-15 / 24-31
            v16bf afrag[4];
            #pragma unroll
            for (int mi = 0; mi < 4; ++mi) {
                int m = waveM * 64 + mi * 16 + l15;
                const unsigned short* p = &As[m * LDT + kk + hi16 * 8];
                v16bf t;
                ((v8bf*)&t)[0] = *(const v8bf*)(p);
                ((v8bf*)&t)[1] = *(const v8bf*)(p + 16);
                afrag[mi] = t;
            }
            // B frags: lane = column; lanes 0-15 K 0-7 / 16-23 ; lanes 16-31 K 8-15 / 24-31
            v16bf bfrag[2];
            #pragma unroll
            for (int ni = 0; ni < 2; ++ni) {
                int n = waveN * 32 + ni * 16 + l15;
                const unsigned short* p = &Bs[n * LDT + kk + hi16 * 8];
                v16bf t;
                ((v8bf*)&t)[0] = *(const v8bf*)(p);
                ((v8bf*)&t)[1] = *(const v8bf*)(p + 16);
                bfrag[ni] = t;
            }
            #pragma unroll
            for (int mi = 0; mi < 4; ++mi)
                #pragma unroll
                for (int ni = 0; ni < 2; ++ni)
                    acc[mi][ni] = __builtin_amdgcn_wmma_f32_16x16x32_bf16(
                        false, afrag[mi], false, bfrag[ni],
                        (short)0, acc[mi][ni], false, false);
        }
    }

    // ---- epilogue: exact fp32 per-output-row scale, coalesced row stores ----
    #pragma unroll
    for (int ni = 0; ni < 2; ++ni) {
        int n = blockN0 + waveN * 32 + ni * 16 + l15;
        float s = scale[n];
        #pragma unroll
        for (int mi = 0; mi < 4; ++mi) {
            int mbase = blockM0 + waveM * 64 + mi * 16 + hi16 * 8;
            #pragma unroll
            for (int r = 0; r < 8; ++r) {
                out[(size_t)(mbase + r) * OUT_N + n] = acc[mi][ni][r] * s;
            }
        }
    }
}

extern "C" void kernel_launch(void* const* d_in, const int* in_sizes, int n_in,
                              void* d_out, int out_size, void* d_ws, size_t ws_size,
                              hipStream_t stream) {
    (void)in_sizes; (void)n_in; (void)out_size; (void)d_ws; (void)ws_size;
    const float* x      = (const float*)d_in[0];
    const int*   packed = (const int*)d_in[1];
    const float* scale  = (const float*)d_in[2];
    float*       out    = (float*)d_out;

    dim3 grid(OUT_N / BN, M_ROWS / BM);   // 86 x 64 blocks
    ternary_gemm_bf16_wmma<<<grid, 256, 0, stream>>>(x, packed, scale, out);
}
